// DynamicsPredictor_9887014715547
// MI455X (gfx1250) — compile-verified, matrix operand order checked by reading, not run on hardware
//
#include <hip/hip_runtime.h>
#include <hip/hip_bf16.h>

// ---------------- problem constants (from reference) ----------------
constexpr int BATCH  = 2;
constexpr int N_P    = 2000;
constexpr int N_S    = 48;
constexpr int NN     = N_P + N_S;   // 2048
constexpr int N_REL  = 8192;
constexpr int N_HIS  = 4;
constexpr int N_INST = 8;
constexpr int NF     = 128;
constexpr int IN_DIM = 19;
constexpr int IN_PAD = 32;
constexpr int REL_DIM = 56;
constexpr int REL_PAD = 64;
constexpr float MOTION_CLAMP = 100.0f;

typedef _Float16 v8h  __attribute__((ext_vector_type(8)));
typedef _Float16 v16h __attribute__((ext_vector_type(16)));
typedef float    v8f  __attribute__((ext_vector_type(8)));

// ---------------- index extraction from one-hot Rr/Rs ----------------
// one wave (32 lanes) per row of 2048; coalesced strided scan + shuffle max
__global__ __launch_bounds__(256)
void k_extract_idx(const float* __restrict__ Rr, const float* __restrict__ Rs,
                   int* __restrict__ rr, int* __restrict__ rs) {
    const int warp = (blockIdx.x * blockDim.x + threadIdx.x) >> 5;
    const int lane = threadIdx.x & 31;
    const int rowsPer = BATCH * N_REL;                 // 16384
    if (warp >= 2 * rowsPer) return;
    const float* src = (warp < rowsPer) ? Rr : Rs;
    int*         dst = (warp < rowsPer) ? rr : rs;
    const int row = (warp < rowsPer) ? warp : warp - rowsPer;
    const float* p = src + (size_t)row * NN;
    int cand = -1;
    #pragma unroll 4
    for (int t = 0; t < NN / 32; ++t) {
        float v = p[t * 32 + lane];
        if (v > 0.5f) cand = t * 32 + lane;
    }
    #pragma unroll
    for (int off = 16; off > 0; off >>= 1) {
        int o = __shfl_xor(cand, off, 32);
        cand = cand > o ? cand : o;
    }
    if (lane == 0) dst[row] = cand;
}

// ---------------- build particle inputs [B,N,19] (+padded f16) ----------------
__global__ __launch_bounds__(256)
void k_build_pinputs(const float* __restrict__ state, const float* __restrict__ attrs,
                     const float* __restrict__ action, const float* __restrict__ den,
                     const float* __restrict__ phys,
                     float* __restrict__ p32, _Float16* __restrict__ p16) {
    const int t = blockIdx.x * blockDim.x + threadIdx.x;
    if (t >= BATCH * NN) return;
    const int b = t / NN, n = t % NN;
    float vals[IN_DIM];
    vals[0] = attrs[(size_t)t * 2 + 0];
    vals[1] = attrs[(size_t)t * 2 + 1];
    // temporal residuals + last state, (b,n,h,3) flatten
    #pragma unroll
    for (int h = 0; h < 3; ++h)
        #pragma unroll
        for (int d = 0; d < 3; ++d)
            vals[2 + h * 3 + d] =
                state[(((size_t)b * 4 + h + 1) * NN + n) * 3 + d] -
                state[(((size_t)b * 4 + h) * NN + n) * 3 + d];
    #pragma unroll
    for (int d = 0; d < 3; ++d)
        vals[2 + 9 + d] = state[(((size_t)b * 4 + 3) * NN + n) * 3 + d];
    vals[14] = (n < N_P) ? phys[b] : 0.f;        // physics_param [B,1]
    vals[15] = action[(size_t)t * 3 + 0];
    vals[16] = action[(size_t)t * 3 + 1];
    vals[17] = action[(size_t)t * 3 + 2];
    vals[18] = (n < N_P) ? den[b] : 0.f;
    float*    o32 = p32 + (size_t)t * IN_DIM;
    _Float16* o16 = p16 + (size_t)t * IN_PAD;
    #pragma unroll
    for (int i = 0; i < IN_DIM; ++i) { o32[i] = vals[i]; o16[i] = (_Float16)vals[i]; }
    #pragma unroll
    for (int i = IN_DIM; i < IN_PAD; ++i) o16[i] = (_Float16)0.f;
}

// ---------------- build relation inputs [B,n_rel,56] -> f16 padded 64 ----------------
__global__ __launch_bounds__(256)
void k_build_relinputs(const int* __restrict__ rr, const int* __restrict__ rs,
                       const float* __restrict__ p32, const float* __restrict__ attrs,
                       const float* __restrict__ pinst, _Float16* __restrict__ rel16) {
    const int t = blockIdx.x * blockDim.x + threadIdx.x;
    if (t >= BATCH * N_REL) return;
    const int b = t / N_REL;
    const int ir = rr[t], is = rs[t];
    const float* pr = p32 + (size_t)(b * NN + ir) * IN_DIM;
    const float* ps = p32 + (size_t)(b * NN + is) * IN_DIM;
    _Float16* o = rel16 + (size_t)t * REL_PAD;
    #pragma unroll
    for (int i = 0; i < IN_DIM; ++i) o[i] = (_Float16)pr[i];
    #pragma unroll
    for (int i = 0; i < IN_DIM; ++i) o[IN_DIM + i] = (_Float16)ps[i];
    o[38] = (_Float16)attrs[(size_t)(b * NN + ir) * 2 + 0];
    o[39] = (_Float16)attrs[(size_t)(b * NN + ir) * 2 + 1];
    o[40] = (_Float16)attrs[(size_t)(b * NN + is) * 2 + 0];
    o[41] = (_Float16)attrs[(size_t)(b * NN + is) * 2 + 1];
    float gd = 0.f;
    #pragma unroll
    for (int k = 0; k < N_INST; ++k) {
        float gr = (ir < N_P) ? pinst[((size_t)b * N_P + ir) * N_INST + k] : 0.f;
        float gs = (is < N_P) ? pinst[((size_t)b * N_P + is) * N_INST + k] : 0.f;
        gd += fabsf(gr - gs);
    }
    o[42] = (_Float16)gd;
    #pragma unroll
    for (int i = 0; i < 12; ++i) o[43 + i] = (_Float16)(pr[2 + i] - ps[2 + i]); // pos_diff
    o[55] = (_Float16)(pr[18] - ps[18]);                                        // dens_diff
    #pragma unroll
    for (int i = REL_DIM; i < REL_PAD; ++i) o[i] = (_Float16)0.f;
}

// ---------------- weight convert+transpose: fp32 [K,128] -> f16 [128,Kpad] ----------------
__global__ __launch_bounds__(256)
void k_wtrans(const float* __restrict__ W, _Float16* __restrict__ Wt, int K, int Kpad) {
    const int t = blockIdx.x * blockDim.x + threadIdx.x;
    if (t >= NF * Kpad) return;
    const int n = t / Kpad, k = t % Kpad;
    Wt[t] = (k < K) ? (_Float16)W[(size_t)k * NF + n] : (_Float16)0.f;
}

// ---------------- WMMA GEMM: A[M x Kpad](f16) @ Bt[128 x Kpad](f16)^T + bias ----------------
// 8 waves per block; wave w computes the 16x16 tile at (blockIdx.x*16, w*16). N fixed = 128.
__global__ __launch_bounds__(256)
void k_gemm(const _Float16* __restrict__ A, const _Float16* __restrict__ Bt,
            const float* __restrict__ bias, const float* __restrict__ resid,
            float* __restrict__ outF, _Float16* __restrict__ outH,
            int Kpad, int relu) {
    const int lane = threadIdx.x & 31;
    const int wave = threadIdx.x >> 5;
    const int m0 = blockIdx.x * 16;
    const int n0 = wave * 16;
    const int rsel = lane & 15;        // A-row / B-col within tile
    const int kb   = (lane >> 4) * 8;  // 16-bit operand K sub-offset (ISA layout)
    const _Float16* arow = A  + (size_t)(m0 + rsel) * Kpad;
    const _Float16* bcol = Bt + (size_t)(n0 + rsel) * Kpad;
    v8f acc = {};
    for (int kc = 0; kc < Kpad; kc += 32) {
        v8h al = *(const v8h*)(arow + kc + kb);
        v8h ah = *(const v8h*)(arow + kc + 16 + kb);
        v8h bl = *(const v8h*)(bcol + kc + kb);
        v8h bh = *(const v8h*)(bcol + kc + 16 + kb);
        v16h a = __builtin_shufflevector(al, ah, 0,1,2,3,4,5,6,7,8,9,10,11,12,13,14,15);
        v16h b = __builtin_shufflevector(bl, bh, 0,1,2,3,4,5,6,7,8,9,10,11,12,13,14,15);
        acc = __builtin_amdgcn_wmma_f32_16x16x32_f16(false, a, false, b,
                                                     (short)0, acc, false, false);
    }
    const int col = n0 + rsel;
    const float bv = bias ? bias[col] : 0.f;
    const int rowbase = m0 + ((lane >> 4) * 8);   // C/D layout: lanes 16-31 hold M=8..15
    #pragma unroll
    for (int r = 0; r < 8; ++r) {
        const int row = rowbase + r;
        float v = acc[r] + bv;
        if (resid) v += resid[(size_t)row * NF + col];
        if (relu)  v = fmaxf(v, 0.f);
        if (outF) outF[(size_t)row * NF + col] = v;
        if (outH) outH[(size_t)row * NF + col] = (_Float16)v;
    }
}

// ---------------- helpers ----------------
__global__ __launch_bounds__(256)
void k_zero(float* __restrict__ p, int n) {
    const int t = blockIdx.x * blockDim.x + threadIdx.x;
    if (t < n) p[t] = 0.f;
}

__global__ __launch_bounds__(256)
void k_copyh(_Float16* __restrict__ dst, const _Float16* __restrict__ src, int n) {
    const int t = blockIdx.x * blockDim.x + threadIdx.x;
    if (t < n) dst[t] = src[t];
}

// scatter-add: agg[b, rr[e], :] += effect_rel[e, :]
__global__ __launch_bounds__(256)
void k_scatter(float* __restrict__ agg, const float* __restrict__ effrel,
               const int* __restrict__ rr) {
    const int t = blockIdx.x * blockDim.x + threadIdx.x;
    if (t >= BATCH * N_REL * NF) return;
    const int e = t / NF, c = t % NF;
    const int b = e / N_REL;
    atomicAdd(&agg[((size_t)b * NN + rr[e]) * NF + c], effrel[(size_t)e * NF + c]);
}

// rel_in = [relation_encode | eff_r | eff_s]  (f16, [16384 x 384])
__global__ __launch_bounds__(256)
void k_build_relcat(_Float16* __restrict__ relcat, const _Float16* __restrict__ relenc,
                    const _Float16* __restrict__ peff16,
                    const int* __restrict__ rr, const int* __restrict__ rs) {
    const int t = blockIdx.x * blockDim.x + threadIdx.x;
    if (t >= BATCH * N_REL * NF) return;
    const int e = t / NF, c = t % NF;
    const int b = e / N_REL;
    _Float16* o = relcat + (size_t)e * (3 * NF);
    o[c]            = relenc[(size_t)e * NF + c];
    o[NF + c]       = peff16[((size_t)b * NN + rr[e]) * NF + c];
    o[2 * NF + c]   = peff16[((size_t)b * NN + rs[e]) * NF + c];
}

// p_in = [particle_encode | agg]  (f16, [4096 x 256])
__global__ __launch_bounds__(256)
void k_build_pin(_Float16* __restrict__ pin, const _Float16* __restrict__ pe16,
                 const float* __restrict__ agg) {
    const int t = blockIdx.x * blockDim.x + threadIdx.x;
    if (t >= BATCH * NN * NF) return;
    const int r = t / NF, c = t % NF;
    pin[(size_t)r * 2 * NF + c]      = pe16[t];
    pin[(size_t)r * 2 * NF + NF + c] = (_Float16)agg[t];
}

// gather first n_p rows of each batch for the predictor
__global__ __launch_bounds__(256)
void k_gather_np(_Float16* __restrict__ dst, const _Float16* __restrict__ peff16) {
    const int t = blockIdx.x * blockDim.x + threadIdx.x;
    if (t >= BATCH * N_P * NF) return;
    const int row = t / NF, c = t % NF;
    const int b = row / N_P, i = row % N_P;
    dst[t] = peff16[((size_t)b * NN + i) * NF + c];
}

// final 128->3 layer + clamp + add last state
__global__ __launch_bounds__(256)
void k_final(float* __restrict__ out, const _Float16* __restrict__ h,
             const float* __restrict__ W2, const float* __restrict__ b2,
             const float* __restrict__ state) {
    const int t = blockIdx.x * blockDim.x + threadIdx.x;
    if (t >= BATCH * N_P * 3) return;
    const int row = t / 3, j = t % 3;
    const int b = row / N_P, i = row % N_P;
    float acc = b2[j];
    const _Float16* hr = h + (size_t)row * NF;
    #pragma unroll 8
    for (int k = 0; k < NF; ++k) acc += (float)hr[k] * W2[(size_t)k * 3 + j];
    acc = fminf(fmaxf(acc, -MOTION_CLAMP), MOTION_CLAMP);
    out[t] = state[(((size_t)b * 4 + 3) * NN + i) * 3 + j] + acc;
}

// ---------------- launch ----------------
extern "C" void kernel_launch(void* const* d_in, const int* in_sizes, int n_in,
                              void* d_out, int out_size, void* d_ws, size_t ws_size,
                              hipStream_t stream) {
    const float* state  = (const float*)d_in[0];
    const float* attrs  = (const float*)d_in[1];
    const float* Rr     = (const float*)d_in[2];
    const float* Rs     = (const float*)d_in[3];
    const float* pinst  = (const float*)d_in[4];
    const float* action = (const float*)d_in[5];
    const float* pden   = (const float*)d_in[6];
    const float* phys   = (const float*)d_in[7];
    const float* pe_W[3] = {(const float*)d_in[8],  (const float*)d_in[10], (const float*)d_in[12]};
    const float* pe_b[3] = {(const float*)d_in[9],  (const float*)d_in[11], (const float*)d_in[13]};
    const float* re_W[3] = {(const float*)d_in[14], (const float*)d_in[16], (const float*)d_in[18]};
    const float* re_b[3] = {(const float*)d_in[15], (const float*)d_in[17], (const float*)d_in[19]};
    const float* pp_W = (const float*)d_in[20];  const float* pp_b = (const float*)d_in[21];
    const float* rp_W = (const float*)d_in[22];  const float* rp_b = (const float*)d_in[23];
    const float* np_W0 = (const float*)d_in[24]; const float* np_b0 = (const float*)d_in[25];
    const float* np_W1 = (const float*)d_in[26]; const float* np_b1 = (const float*)d_in[27];
    const float* np_W2 = (const float*)d_in[28]; const float* np_b2 = (const float*)d_in[29];
    float* out = (float*)d_out;

    // bump allocator over workspace
    char* base = (char*)d_ws;
    size_t off = 0;
    auto alloc = [&](size_t bytes) -> void* {
        off = (off + 255) & ~(size_t)255;
        void* r = base + off;
        off += bytes;
        return r;
    };
    const int ROWS_E = BATCH * N_REL;   // 16384 edge rows
    const int ROWS_N = BATCH * NN;      // 4096 node rows
    const int ROWS_Q = BATCH * N_P;     // 4000 predictor rows

    int*      rr_idx = (int*)alloc(sizeof(int) * ROWS_E);
    int*      rs_idx = (int*)alloc(sizeof(int) * ROWS_E);
    float*    p32    = (float*)alloc(sizeof(float) * ROWS_N * IN_DIM);
    _Float16* Xp16   = (_Float16*)alloc(2 * (size_t)ROWS_N * IN_PAD);
    _Float16* rel16  = (_Float16*)alloc(2 * (size_t)ROWS_E * REL_PAD);
    _Float16* WpeT0  = (_Float16*)alloc(2 * NF * IN_PAD);
    _Float16* WpeT1  = (_Float16*)alloc(2 * NF * NF);
    _Float16* WpeT2  = (_Float16*)alloc(2 * NF * NF);
    _Float16* WreT0  = (_Float16*)alloc(2 * NF * REL_PAD);
    _Float16* WreT1  = (_Float16*)alloc(2 * NF * NF);
    _Float16* WreT2  = (_Float16*)alloc(2 * NF * NF);
    _Float16* WppT   = (_Float16*)alloc(2 * NF * 2 * NF);
    _Float16* WrpT   = (_Float16*)alloc(2 * NF * 3 * NF);
    _Float16* WnpT0  = (_Float16*)alloc(2 * NF * NF);
    _Float16* WnpT1  = (_Float16*)alloc(2 * NF * NF);
    _Float16* bufA   = (_Float16*)alloc(2 * (size_t)ROWS_E * NF);
    _Float16* bufB   = (_Float16*)alloc(2 * (size_t)ROWS_E * NF);
    _Float16* relenc = (_Float16*)alloc(2 * (size_t)ROWS_E * NF);
    _Float16* pe16   = (_Float16*)alloc(2 * (size_t)ROWS_N * NF);
    float*    peff   = (float*)alloc(sizeof(float) * ROWS_N * NF);
    _Float16* peff16 = (_Float16*)alloc(2 * (size_t)ROWS_N * NF);
    _Float16* relcat = (_Float16*)alloc(2 * (size_t)ROWS_E * 3 * NF);
    float*    effrel = (float*)alloc(sizeof(float) * ROWS_E * NF);
    float*    agg    = (float*)alloc(sizeof(float) * ROWS_N * NF);
    _Float16* pin16  = (_Float16*)alloc(2 * (size_t)ROWS_N * 2 * NF);
    _Float16* np16a  = (_Float16*)alloc(2 * (size_t)ROWS_Q * NF);
    _Float16* np16b  = (_Float16*)alloc(2 * (size_t)ROWS_Q * NF);
    (void)ws_size; (void)in_sizes; (void)n_in; (void)out_size;

    auto cdiv = [](int a, int b) { return (a + b - 1) / b; };

    // 1) recover edge indices from one-hot Rr/Rs
    k_extract_idx<<<cdiv(2 * ROWS_E * 32, 256), 256, 0, stream>>>(Rr, Rs, rr_idx, rs_idx);

    // 2) convert + transpose weights to f16 [128 x Kpad]
    k_wtrans<<<cdiv(NF * IN_PAD, 256), 256, 0, stream>>>(pe_W[0], WpeT0, IN_DIM, IN_PAD);
    k_wtrans<<<cdiv(NF * NF, 256), 256, 0, stream>>>(pe_W[1], WpeT1, NF, NF);
    k_wtrans<<<cdiv(NF * NF, 256), 256, 0, stream>>>(pe_W[2], WpeT2, NF, NF);
    k_wtrans<<<cdiv(NF * REL_PAD, 256), 256, 0, stream>>>(re_W[0], WreT0, REL_DIM, REL_PAD);
    k_wtrans<<<cdiv(NF * NF, 256), 256, 0, stream>>>(re_W[1], WreT1, NF, NF);
    k_wtrans<<<cdiv(NF * NF, 256), 256, 0, stream>>>(re_W[2], WreT2, NF, NF);
    k_wtrans<<<cdiv(NF * 2 * NF, 256), 256, 0, stream>>>(pp_W, WppT, 2 * NF, 2 * NF);
    k_wtrans<<<cdiv(NF * 3 * NF, 256), 256, 0, stream>>>(rp_W, WrpT, 3 * NF, 3 * NF);
    k_wtrans<<<cdiv(NF * NF, 256), 256, 0, stream>>>(np_W0, WnpT0, NF, NF);
    k_wtrans<<<cdiv(NF * NF, 256), 256, 0, stream>>>(np_W1, WnpT1, NF, NF);

    // 3) build particle + relation inputs
    k_build_pinputs<<<cdiv(ROWS_N, 256), 256, 0, stream>>>(state, attrs, action, pden, phys,
                                                           p32, Xp16);
    k_build_relinputs<<<cdiv(ROWS_E, 256), 256, 0, stream>>>(rr_idx, rs_idx, p32, attrs,
                                                             pinst, rel16);

    // 4) particle encoder (3x Linear+ReLU), WMMA f16->f32
    k_gemm<<<ROWS_N / 16, 256, 0, stream>>>(Xp16, WpeT0, pe_b[0], nullptr, nullptr, bufA, IN_PAD, 1);
    k_gemm<<<ROWS_N / 16, 256, 0, stream>>>(bufA, WpeT1, pe_b[1], nullptr, nullptr, bufB, NF, 1);
    k_gemm<<<ROWS_N / 16, 256, 0, stream>>>(bufB, WpeT2, pe_b[2], nullptr, peff, pe16, NF, 1);
    k_copyh<<<cdiv(ROWS_N * NF, 256), 256, 0, stream>>>(peff16, pe16, ROWS_N * NF);

    // 5) relation encoder
    k_gemm<<<ROWS_E / 16, 256, 0, stream>>>(rel16, WreT0, re_b[0], nullptr, nullptr, bufA, REL_PAD, 1);
    k_gemm<<<ROWS_E / 16, 256, 0, stream>>>(bufA, WreT1, re_b[1], nullptr, nullptr, bufB, NF, 1);
    k_gemm<<<ROWS_E / 16, 256, 0, stream>>>(bufB, WreT2, re_b[2], nullptr, nullptr, relenc, NF, 1);

    // 6) propagation: 3 steps of relation->particle message passing
    for (int step = 0; step < 3; ++step) {
        k_build_relcat<<<cdiv(ROWS_E * NF, 256), 256, 0, stream>>>(relcat, relenc, peff16,
                                                                   rr_idx, rs_idx);
        k_gemm<<<ROWS_E / 16, 256, 0, stream>>>(relcat, WrpT, rp_b, nullptr, effrel, nullptr,
                                                3 * NF, 1);
        k_zero<<<cdiv(ROWS_N * NF, 256), 256, 0, stream>>>(agg, ROWS_N * NF);
        k_scatter<<<cdiv(ROWS_E * NF, 256), 256, 0, stream>>>(agg, effrel, rr_idx);
        k_build_pin<<<cdiv(ROWS_N * NF, 256), 256, 0, stream>>>(pin16, pe16, agg);
        // residual propagator: relu(pin @ ppW + b + peff), in-place on peff/peff16
        k_gemm<<<ROWS_N / 16, 256, 0, stream>>>(pin16, WppT, pp_b, peff, peff, peff16, 2 * NF, 1);
    }

    // 7) non-rigid predictor on first n_p rows per batch
    k_gather_np<<<cdiv(ROWS_Q * NF, 256), 256, 0, stream>>>(np16a, peff16);
    k_gemm<<<ROWS_Q / 16, 256, 0, stream>>>(np16a, WnpT0, np_b0, nullptr, nullptr, np16b, NF, 1);
    k_gemm<<<ROWS_Q / 16, 256, 0, stream>>>(np16b, WnpT1, np_b1, nullptr, nullptr, np16a, NF, 1);
    k_final<<<cdiv(ROWS_Q * 3, 256), 256, 0, stream>>>(out, np16a, np_W2, np_b2, state);
}
